// PerformerAttention_76055280878079
// MI455X (gfx1250) — matrix operand order repaired and checked by reading, not compiled, not run on hardware
//
#include <hip/hip_runtime.h>
#include <hip/hip_bf16.h>

typedef _Float16 half_t;
typedef __attribute__((ext_vector_type(16))) _Float16 v16h;
typedef __attribute__((ext_vector_type(8)))  float    v8f;
typedef __attribute__((ext_vector_type(4)))  unsigned u32x4;
typedef __attribute__((ext_vector_type(4)))  int      i32x4;
typedef __attribute__((ext_vector_type(8)))  int      i32x8;

#define HEADS    8
#define DIM_HEAD 64
#define NB_FEAT  256
#define SEQ      2048
#define BATCH    4
#define BHN      (BATCH*HEADS)
#define CHUNKSZ  128
#define NCHUNK   (SEQ/CHUNKSZ)

#define NORMC  0.35355339059327373f   /* 64^-0.25 */
#define RATIO  0.0625f                /* 256^-0.5 */
#define EPSK   1e-4f
#define EPSD   1e-6f

/* LDS strides (elements), padded to dodge bank conflicts */
#define LQ 264   /* 256+8, f16 rows of qp/kp/ShT */
#define LV 136   /* 128+8, f16 rows of vT/scores */

__device__ __forceinline__ v8f wmma_f16(v16h a, v16h b, v8f c) {
  return __builtin_amdgcn_wmma_f32_16x16x32_f16(
      false, a, false, b, (short)0, c, false, false);
}

/* A tile (16x32 f16): lane<16 holds row lane, K = seg..seg+7 and seg+16..seg+23 */
__device__ __forceinline__ v16h lds_A16(const half_t* p) {
  v16h a;
  ((uint4*)&a)[0] = ((const uint4*)p)[0];
  ((uint4*)&a)[1] = ((const uint4*)(p + 16))[0];
  return a;
}
/* B tile (32x16 f16) from [n][k]-major storage: lane<16 holds K = seg..seg+7 and seg+16..seg+23 */
__device__ __forceinline__ v16h lds_B16(const half_t* p) {
  v16h b;
  ((uint4*)&b)[0] = ((const uint4*)p)[0];
  ((uint4*)&b)[1] = ((const uint4*)(p + 16))[0];
  return b;
}
/* A tile from [k][m]-major storage (strided gather) */
__device__ __forceinline__ v16h lds_A_trans(const half_t* base, int ld, int koff,
                                            int m, int seg) {
  v16h a;
  #pragma unroll
  for (int j = 0; j < 8; ++j) {
    a[j]     = base[(koff + seg + j) * ld + m];
    a[j + 8] = base[(koff + seg + 16 + j) * ld + m];
  }
  return a;
}
/* A tile from row-major f32 global, scaled + f16 convert, accumulating sum(x^2) */
__device__ __forceinline__ v16h load_A_global(const float* base, int ldr, int lane,
                                              int kst, float scale, float* ss) {
  int row = lane & 15;
  int seg = (lane < 16) ? 0 : 8;
  const float* p = base + (long)row * ldr + kst * 32 + seg;
  v16h a;
  float s = *ss;
  #pragma unroll
  for (int j = 0; j < 8; ++j) { float x = p[j] * scale;      s += x * x; a[j]     = (half_t)x; }
  #pragma unroll
  for (int j = 0; j < 8; ++j) { float x = p[j + 16] * scale; s += x * x; a[j + 8] = (half_t)x; }
  *ss = s;
  return a;
}
/* B tile from projection f16 [256][64] row-major (K = seg..seg+7, seg+16..seg+23 per lane) */
__device__ __forceinline__ v16h load_B_proj(const half_t* proj16, int m0, int lane, int kst) {
  int n  = lane & 15;
  int kb = kst * 32 + ((lane < 16) ? 0 : 8);
  const half_t* p = proj16 + (m0 + n) * 64 + kb;
  v16h b;
  ((uint4*)&b)[0] = ((const uint4*)p)[0];
  ((uint4*)&b)[1] = ((const uint4*)(p + 16))[0];
  return b;
}

/* ordered-uint encoding for float atomicMax */
__device__ __forceinline__ unsigned fenc(float x) {
  unsigned b = __float_as_uint(x);
  return (b & 0x80000000u) ? ~b : (b | 0x80000000u);
}
__device__ __forceinline__ float fdec(unsigned u) {
  unsigned b = (u & 0x80000000u) ? (u & 0x7fffffffu) : ~u;
  return __uint_as_float(b);
}

/* TDM 2D tile load: rows x cols f16, row-major global (stride elems), LDS rows
   padded by pad_amount DWORDs every row (pad_interval must match row bytes). */
__device__ __forceinline__ void tdm_load_2d_f16(void* lds_dst, const void* gsrc,
                                                unsigned rows, unsigned cols,
                                                unsigned stride_elems,
                                                unsigned pad_en,
                                                unsigned pad_interval_code,
                                                unsigned pad_amount_code) {
  unsigned lds_off = (unsigned)(uintptr_t)lds_dst;
  unsigned long long ga = (unsigned long long)(uintptr_t)gsrc;
  u32x4 g0;
  g0[0] = 1u;                                   /* count=1, user descriptor   */
  g0[1] = lds_off;                              /* lds_addr                   */
  g0[2] = (unsigned)(ga & 0xffffffffu);         /* global_addr[31:0]          */
  g0[3] = (unsigned)((ga >> 32) & 0x01ffffffu) | (2u << 30);  /* type=2       */
  unsigned flags = (1u << 16)                   /* data_size = 2 bytes        */
                 | (pad_en << 20)
                 | (pad_interval_code << 22)
                 | (pad_amount_code << 25);
  unsigned long long q0 = (unsigned long long)flags;      /* atomic_bar=0     */
  unsigned long long td0 = cols;                          /* tensor_dim0      */
  q0 |= (td0 & 0xffffull) << 48;
  unsigned long long q1 = (td0 >> 16) & 0xffffull;
  q1 |= ((unsigned long long)rows) << 16;                 /* tensor_dim1      */
  q1 |= ((unsigned long long)(cols & 0xffffu)) << 48;     /* tile_dim0        */
  unsigned long long q2 = (unsigned long long)(rows & 0xffffu);  /* tile_dim1 */
  unsigned long long st0 = stride_elems;                  /* dim0 stride      */
  q2 |= (st0 & 0xffffffffull) << 32;
  unsigned long long q3 = (st0 >> 32) & 0xffffull;        /* dim1 stride = 0  */
  union { unsigned long long q[4]; i32x8 v; } g1u;
  g1u.q[0] = q0; g1u.q[1] = q1; g1u.q[2] = q2; g1u.q[3] = q3;
  i32x4 zero4 = {0, 0, 0, 0};
#if __clang_major__ >= 23
  i32x8 zero8 = {0, 0, 0, 0, 0, 0, 0, 0};
  __builtin_amdgcn_tensor_load_to_lds(g0, g1u.v, zero4, zero4, zero8, 0);
#else
  __builtin_amdgcn_tensor_load_to_lds(g0, g1u.v, zero4, zero4, 0);
#endif
}
__device__ __forceinline__ void tdm_wait() {
  __builtin_amdgcn_s_wait_tensorcnt(0);
}

/* ---------------- K0: convert projection + permute/convert V, init kmax ------------- */
__global__ void k_prep(const float* __restrict__ v, const float* __restrict__ proj,
                       half_t* __restrict__ v16, half_t* __restrict__ proj16,
                       unsigned* __restrict__ kmax) {
  long gid = (long)blockIdx.x * blockDim.x + threadIdx.x;
  if (gid < NB_FEAT * DIM_HEAD) proj16[gid] = (half_t)proj[gid];
  if (gid < BHN) kmax[gid] = 0u;
  const long TOT = (long)BHN * SEQ * DIM_HEAD;
  if (gid < TOT) {
    int e  = (int)(gid & 63);
    int n  = (int)((gid >> 6) & (SEQ - 1));
    int bh = (int)(gid >> 17);
    int b = bh >> 3, h = bh & 7;
    v16[gid] = (half_t)v[((long)(b * SEQ + n)) * 512 + h * 64 + e];
  }
}

/* ---------------- K1: per-(b,h) max of key data_dash via WMMA ----------------------- */
__global__ void k_kmax(const float* __restrict__ key, const half_t* __restrict__ proj16,
                       unsigned* __restrict__ kmax) {
  int lane = threadIdx.x & 31, w = threadIdx.x >> 5;
  int bh = blockIdx.x >> 4;
  int n0 = ((blockIdx.x & 15) << 7) + (w << 4);
  int b = bh >> 3, h = bh & 7;
  const float* base = key + ((long)(b * SEQ + n0)) * 512 + h * 64;
  float ss = 0.f;
  v16h a0 = load_A_global(base, 512, lane, 0, NORMC, &ss);
  v16h a1 = load_A_global(base, 512, lane, 1, NORMC, &ss);
  float mx = -3.0e38f;
  for (int ct = 0; ct < 16; ++ct) {
    v8f c = {};
    c = wmma_f16(a0, load_B_proj(proj16, ct * 16, lane, 0), c);
    c = wmma_f16(a1, load_B_proj(proj16, ct * 16, lane, 1), c);
    #pragma unroll
    for (int i = 0; i < 8; ++i) mx = fmaxf(mx, c[i]);
  }
  for (int msk = 16; msk >= 1; msk >>= 1) mx = fmaxf(mx, __shfl_xor(mx, msk));
  __shared__ float red[8];
  if (lane == 0) red[w] = mx;
  __syncthreads();
  if (threadIdx.x == 0) {
    float r = red[0];
    #pragma unroll
    for (int i = 1; i < 8; ++i) r = fmaxf(r, red[i]);
    atomicMax(kmax + bh, fenc(r));
  }
}

/* ---------------- K2: query feature map (per-row max) ------------------------------- */
__global__ void k_qfeat(const float* __restrict__ query, const half_t* __restrict__ proj16,
                        half_t* __restrict__ qp) {
  extern __shared__ char smem[];
  float* ts     = (float*)smem;              /* 8 waves x [16][LQ] f32 */
  float* diag_s = ts + 8 * 16 * LQ;          /* 8 x 16 */
  int lane = threadIdx.x & 31, w = threadIdx.x >> 5;
  int bh = blockIdx.x >> 4;
  int n0 = ((blockIdx.x & 15) << 7) + (w << 4);
  int b = bh >> 3, h = bh & 7;
  float* tw = ts + w * 16 * LQ;
  const float* base = query + ((long)(b * SEQ + n0)) * 512 + h * 64;
  float ss = 0.f;
  v16h a0 = load_A_global(base, 512, lane, 0, NORMC, &ss);
  v16h a1 = load_A_global(base, 512, lane, 1, NORMC, &ss);
  float diag = 0.5f * (ss + __shfl_xor(ss, 16));
  if (lane < 16) diag_s[w * 16 + lane] = diag;
  int g8 = (lane < 16) ? 0 : 8;
  int n15 = lane & 15;
  for (int ct = 0; ct < 16; ++ct) {
    v8f c = {};
    c = wmma_f16(a0, load_B_proj(proj16, ct * 16, lane, 0), c);
    c = wmma_f16(a1, load_B_proj(proj16, ct * 16, lane, 1), c);
    #pragma unroll
    for (int i = 0; i < 8; ++i) tw[(i + g8) * LQ + ct * 16 + n15] = c[i];
  }
  __syncthreads();
  int r  = lane & 15;
  int ch = (lane < 16) ? 0 : 128;
  float mx = -3.0e38f;
  for (int cI = 0; cI < 128; ++cI) mx = fmaxf(mx, tw[r * LQ + ch + cI]);
  mx = fmaxf(mx, __shfl_xor(mx, 16));
  float dg = diag_s[w * 16 + r];
  half_t* outp = qp + ((long)bh * SEQ + n0 + r) * 256 + ch;
  for (int cI = 0; cI < 128; ++cI) {
    float val = tw[r * LQ + ch + cI];
    outp[cI] = (half_t)(RATIO * (__expf(val - dg - mx) + EPSK));
  }
}

/* ---------------- K3: key feature map (global max) ---------------------------------- */
__global__ void k_kfeat(const float* __restrict__ key, const half_t* __restrict__ proj16,
                        const unsigned* __restrict__ kmax, half_t* __restrict__ kp) {
  __shared__ float diag_s[8][16];
  int lane = threadIdx.x & 31, w = threadIdx.x >> 5;
  int bh = blockIdx.x >> 4;
  int n0 = ((blockIdx.x & 15) << 7) + (w << 4);
  int b = bh >> 3, h = bh & 7;
  const float* base = key + ((long)(b * SEQ + n0)) * 512 + h * 64;
  float ss = 0.f;
  v16h a0 = load_A_global(base, 512, lane, 0, NORMC, &ss);
  v16h a1 = load_A_global(base, 512, lane, 1, NORMC, &ss);
  float diag = 0.5f * (ss + __shfl_xor(ss, 16));
  if (lane < 16) diag_s[w][lane] = diag;
  __syncthreads();
  float m = fdec(kmax[bh]);
  int g8 = (lane < 16) ? 0 : 8;
  int n15 = lane & 15;
  for (int ct = 0; ct < 16; ++ct) {
    v8f c = {};
    c = wmma_f16(a0, load_B_proj(proj16, ct * 16, lane, 0), c);
    c = wmma_f16(a1, load_B_proj(proj16, ct * 16, lane, 1), c);
    #pragma unroll
    for (int i = 0; i < 8; ++i) {
      int rr = i + g8;
      float val = RATIO * (__expf(c[i] - diag_s[w][rr] - m) + EPSK);
      kp[((long)bh * SEQ + n0 + rr) * 256 + ct * 16 + n15] = (half_t)val;
    }
  }
}

/* ---------------- K4a: per-chunk state  S_ck = kp^T . v,  z_ck = colsum(kp) --------- */
__global__ void __launch_bounds__(256, 1)
k_state(const half_t* __restrict__ kp, const half_t* __restrict__ v16,
        float* __restrict__ Sc, float* __restrict__ zc) {
  extern __shared__ char smem[];
  half_t* kp_s = (half_t*)smem;          /* [128][LQ] */
  half_t* vT_s = kp_s + 128 * LQ;        /* [64][LV]  vT[e][k] */
  int tid = threadIdx.x, lane = tid & 31, w = tid >> 5;
  int bh = blockIdx.x >> 4, ck = blockIdx.x & 15;
  const half_t* kp_g = kp  + ((long)bh * SEQ + ck * 128) * 256;
  const half_t* v_g  = v16 + ((long)bh * SEQ + ck * 128) * 64;

  if (w == 0) tdm_load_2d_f16(kp_s, kp_g, 128, 256, 256, 1, 6, 3);
  for (int i = tid; i < 128 * 64; i += 256) {
    int kk = i >> 6, e = i & 63;
    vT_s[e * LV + kk] = v_g[i];
  }
  if (w == 0) tdm_wait();
  __syncthreads();

  int g8  = (lane < 16) ? 0 : 8;
  int n15 = lane & 15;
  int sA  = (lane < 16) ? 0 : 8;
  int sB  = (lane < 16) ? 0 : 8;
  float* Sck = Sc + ((long)bh * NCHUNK + ck) * 256 * 64;

  #pragma unroll
  for (int mt2 = 0; mt2 < 2; ++mt2) {
    int mt = w * 2 + mt2;
    float acc[4][8];
    #pragma unroll
    for (int et = 0; et < 4; ++et)
      #pragma unroll
      for (int i = 0; i < 8; ++i) acc[et][i] = 0.f;
    for (int kst = 0; kst < 4; ++kst) {
      v16h a = lds_A_trans(kp_s, LQ, kst * 32, mt * 16 + n15, sA);
      #pragma unroll
      for (int et = 0; et < 4; ++et) {
        v16h bm = lds_B16(vT_s + (et * 16 + n15) * LV + kst * 32 + sB);
        v8f c;
        #pragma unroll
        for (int i = 0; i < 8; ++i) c[i] = acc[et][i];
        c = wmma_f16(a, bm, c);
        #pragma unroll
        for (int i = 0; i < 8; ++i) acc[et][i] = c[i];
      }
    }
    #pragma unroll
    for (int et = 0; et < 4; ++et)
      #pragma unroll
      for (int i = 0; i < 8; ++i)
        Sck[(mt * 16 + i + g8) * 64 + et * 16 + n15] = acc[et][i];
  }
  float zacc = 0.f;
  for (int j = 0; j < 128; ++j) zacc += (float)kp_s[j * LQ + tid];
  zc[((long)bh * NCHUNK + ck) * 256 + tid] = zacc;
}

/* ---------------- K4b: exclusive prefix over chunk states (in place) ---------------- */
__global__ void k_scan(float* __restrict__ Sc, float* __restrict__ zc) {
  int tid = threadIdx.x;
  int bh = blockIdx.x;
  float* S = Sc + (long)bh * NCHUNK * 256 * 64;
  float* z = zc + (long)bh * NCHUNK * 256;
  float run[64];
  #pragma unroll
  for (int i = 0; i < 64; ++i) run[i] = 0.f;
  float rz = 0.f;
  for (int ck = 0; ck < NCHUNK; ++ck) {
    float* Sck = S + (long)ck * 256 * 64;
    if (ck + 1 < NCHUNK)
      __builtin_prefetch((const char*)(Sck + 256 * 64) + tid * 256, 0, 2);
    #pragma unroll
    for (int i = 0; i < 64; ++i) {
      float cur = Sck[tid + i * 256];
      Sck[tid + i * 256] = run[i];
      run[i] += cur;
    }
    float cz = z[ck * 256 + tid];
    z[ck * 256 + tid] = rz;
    rz += cz;
  }
}

/* ---------------- K4c: per-chunk attention (fully parallel over bh x chunk) --------- */
__global__ void __launch_bounds__(256, 1)
k_attn2(const half_t* __restrict__ qp, const half_t* __restrict__ kp,
        const half_t* __restrict__ v16, const float* __restrict__ Sc,
        const float* __restrict__ zc, float* __restrict__ out) {
  extern __shared__ char smem[];
  half_t* qp_s  = (half_t*)smem;             /* [128][LQ] */
  half_t* kp_s  = qp_s + 128 * LQ;           /* [128][LQ] */
  half_t* vT_s  = kp_s + 128 * LQ;           /* [64][LV]  vT[e][k] */
  half_t* ShT_s = vT_s + 64 * LV;            /* [64][LQ]  ShT[e][m] f16 of prefix S */
  half_t* sc_s  = ShT_s + 64 * LQ;           /* [128][LV] masked scores */
  float*  z_s   = (float*)(sc_s + 128 * LV); /* [256] prefix z */
  float*  den_s = z_s + 256;                 /* [128] rowsum -> 1/denom */
  float*  dq_s  = den_s + 128;               /* [256] per-halfrow partial qd */
  float*  qs_s  = dq_s + 256;                /* [256] per-halfrow partial qs */

  int tid = threadIdx.x, lane = tid & 31, w = tid >> 5;
  int bh = blockIdx.x >> 4, ck = blockIdx.x & 15;
  int b = bh >> 3, h = bh & 7;
  const half_t* qp_g = qp  + ((long)bh * SEQ + ck * 128) * 256;
  const half_t* kp_g = kp  + ((long)bh * SEQ + ck * 128) * 256;
  const half_t* v_g  = v16 + ((long)bh * SEQ + ck * 128) * 64;
  const float*  Sck  = Sc  + ((long)bh * NCHUNK + ck) * 256 * 64;
  const float*  zck  = zc  + ((long)bh * NCHUNK + ck) * 256;

  /* ---- stage: wave0 drives TDM for qp/kp while others stage v / state / z ---- */
  if (w == 0) {
    tdm_load_2d_f16(qp_s, qp_g, 128, 256, 256, 1, 6, 3);
    tdm_load_2d_f16(kp_s, kp_g, 128, 256, 256, 1, 6, 3);
  }
  for (int i = tid; i < 128 * 64; i += 256) {
    int kk = i >> 6, e = i & 63;
    vT_s[e * LV + kk] = v_g[i];
  }
  for (int i = tid; i < 256 * 64; i += 256) {
    int m = i >> 6, e = i & 63;
    ShT_s[e * LQ + m] = (half_t)Sck[i];
  }
  z_s[tid] = zck[tid];
  if (w == 0) tdm_wait();
  __syncthreads();

  int g8  = (lane < 16) ? 0 : 8;
  int n15 = lane & 15;
  int sA  = (lane < 16) ? 0 : 8;
  int sB  = (lane < 16) ? 0 : 8;
  int R0  = w << 4;

  /* P1: scores = qp . kp^T, causal mask, rowsum */
  {
    float acc[8][8];
    #pragma unroll
    for (int ct = 0; ct < 8; ++ct)
      #pragma unroll
      for (int i = 0; i < 8; ++i) acc[ct][i] = 0.f;
    for (int kst = 0; kst < 8; ++kst) {
      v16h a = lds_A16(qp_s + (R0 + n15) * LQ + kst * 32 + sA);
      #pragma unroll
      for (int ct = 0; ct < 8; ++ct) {
        v16h bm = lds_B16(kp_s + (ct * 16 + n15) * LQ + kst * 32 + sB);
        v8f c;
        #pragma unroll
        for (int i = 0; i < 8; ++i) c[i] = acc[ct][i];
        c = wmma_f16(a, bm, c);
        #pragma unroll
        for (int i = 0; i < 8; ++i) acc[ct][i] = c[i];
      }
    }
    float rs[8];
    #pragma unroll
    for (int i = 0; i < 8; ++i) rs[i] = 0.f;
    #pragma unroll
    for (int ct = 0; ct < 8; ++ct) {
      #pragma unroll
      for (int i = 0; i < 8; ++i) {
        int R = R0 + i + g8;
        int Cc = ct * 16 + n15;
        float vsc = (Cc <= R) ? acc[ct][i] : 0.f;
        sc_s[R * LV + Cc] = (half_t)vsc;
        rs[i] += vsc;
      }
    }
    #pragma unroll
    for (int i = 0; i < 8; ++i) {
      float s = rs[i];
      s += __shfl_xor(s, 1); s += __shfl_xor(s, 2);
      s += __shfl_xor(s, 4); s += __shfl_xor(s, 8);
      if (n15 == 0) den_s[R0 + i + g8] = s;
    }
  }
  __syncthreads();

  /* P1b: denominator: 1 / (qp.z_prefix + eps*sum(qp) + rowsum)
     all 256 threads: row = tid&127, m-half = tid>>7 */
  {
    int r = tid & 127, mh = (tid >> 7) * 128;
    float qd = 0.f, qs = 0.f;
    const half_t* qrow = qp_s + r * LQ + mh;
    const float*  zrow = z_s + mh;
    for (int m = 0; m < 128; ++m) {
      float qv = (float)qrow[m];
      qd += qv * zrow[m];
      qs += qv;
    }
    dq_s[tid] = qd;
    qs_s[tid] = qs;
  }
  __syncthreads();
  if (tid < 128) {
    float qd = dq_s[tid] + dq_s[tid + 128];
    float qs = qs_s[tid] + qs_s[tid + 128];
    den_s[tid] = 1.f / (qd + EPSD * qs + den_s[tid]);
  }
  __syncthreads();

  /* P2: out = (qp . S_prefix + scores . v) * dinv */
  {
    float acc[4][8];
    #pragma unroll
    for (int et = 0; et < 4; ++et)
      #pragma unroll
      for (int i = 0; i < 8; ++i) acc[et][i] = 0.f;
    for (int kst = 0; kst < 8; ++kst) {
      v16h a = lds_A16(qp_s + (R0 + n15) * LQ + kst * 32 + sA);
      #pragma unroll
      for (int et = 0; et < 4; ++et) {
        v16h bm = lds_B16(ShT_s + (et * 16 + n15) * LQ + kst * 32 + sB);
        v8f c;
        #pragma unroll
        for (int i = 0; i < 8; ++i) c[i] = acc[et][i];
        c = wmma_f16(a, bm, c);
        #pragma unroll
        for (int i = 0; i < 8; ++i) acc[et][i] = c[i];
      }
    }
    for (int kst = 0; kst < 4; ++kst) {
      v16h a = lds_A16(sc_s + (R0 + n15) * LV + kst * 32 + sA);
      #pragma unroll
      for (int et = 0; et < 4; ++et) {
        v16h bm = lds_B16(vT_s + (et * 16 + n15) * LV + kst * 32 + sB);
        v8f c;
        #pragma unroll
        for (int i = 0; i < 8; ++i) c[i] = acc[et][i];
        c = wmma_f16(a, bm, c);
        #pragma unroll
        for (int i = 0; i < 8; ++i) acc[et][i] = c[i];
      }
    }
    #pragma unroll
    for (int et = 0; et < 4; ++et) {
      #pragma unroll
      for (int i = 0; i < 8; ++i) {
        int R = R0 + i + g8;
        int e = et * 16 + n15;
        out[((long)(b * SEQ + ck * 128 + R)) * 512 + h * 64 + e] = acc[et][i] * den_s[R];
      }
    }
  }
}

extern "C" void kernel_launch(void* const* d_in, const int* in_sizes, int n_in,
                              void* d_out, int out_size, void* d_ws, size_t ws_size,
                              hipStream_t stream) {
  const float* q    = (const float*)d_in[0];
  const float* k    = (const float*)d_in[1];
  const float* v    = (const float*)d_in[2];
  const float* proj = (const float*)d_in[3];
  float* out = (float*)d_out;
  char* ws = (char*)d_ws;

  half_t*   qp     = (half_t*)(ws);                 /* 32 MiB  */
  half_t*   kp     = (half_t*)(ws + 33554432);      /* 32 MiB  */
  half_t*   v16    = (half_t*)(ws + 67108864);      /* 8 MiB   */
  half_t*   proj16 = (half_t*)(ws + 75497472);      /* 32 KiB  */
  unsigned* kmax   = (unsigned*)(ws + 75530240);    /* 512 B   */
  float*    Sc     = (float*)(ws + 75530752);       /* 32 MiB  chunk states  */
  float*    zc     = (float*)(ws + 109085184);      /* 512 KiB chunk z       */

  const size_t SMEM_Q  = (size_t)(8 * 16 * LQ) * 4 + 8 * 16 * 4;
  const size_t SMEM_ST = (size_t)(128 * LQ + 64 * LV) * 2;
  const size_t SMEM_C  = (size_t)(128 * LQ + 128 * LQ + 64 * LV + 64 * LQ + 128 * LV) * 2
                       + (size_t)(256 + 128 + 256 + 256) * 4;

  long tot = (long)BHN * SEQ * DIM_HEAD;
  k_prep <<<(int)((tot + 255) / 256), 256, 0, stream>>>(v, proj, v16, proj16, kmax);
  k_kmax <<<BHN * 16, 256, 0, stream>>>(k, proj16, kmax);
  k_qfeat<<<BHN * 16, 256, SMEM_Q, stream>>>(q, proj16, qp);
  k_kfeat<<<BHN * 16, 256, 0, stream>>>(k, proj16, kmax, kp);
  k_state<<<BHN * NCHUNK, 256, SMEM_ST, stream>>>(kp, v16, Sc, zc);
  k_scan <<<BHN, 256, 0, stream>>>(Sc, zc);
  k_attn2<<<BHN * NCHUNK, 256, SMEM_C, stream>>>(qp, kp, v16, Sc, zc, out);
}
